// GVGRU1_76716705841839
// MI455X (gfx1250) — matrix operand order repaired and checked by reading, not compiled, not run on hardware
//
#include <hip/hip_runtime.h>
#include <hip/hip_bf16.h>
#include <math.h>

// ---------------------------------------------------------------------------
// GVGRU: reversed-time gated recurrence.  B=1024, T=128, H=768.
// 64 persistent workgroups; each owns a 16-row batch tile for all 128 steps.
// Both per-step GEMMs -> v_wmma_f32_16x16x32_bf16, fp32 accumulation.
// Weights pre-converted to bf16 once into d_ws; streamed each step.
// WGP-scope global_prefetch_b8 (inline asm, scope field = 0) pulls the next-but-one
// k-block's weight rows into the WGP cache so the b128 loads hit near cache
// instead of paying L2 latency.  (SE-scope prefetch from __builtin_prefetch
// only touches GL2, where the 2.4 MB of weights already live.)
// ---------------------------------------------------------------------------

typedef __attribute__((ext_vector_type(16))) __bf16 v16bf;
typedef __attribute__((ext_vector_type(8)))  __bf16 v8bf;
typedef __attribute__((ext_vector_type(8)))  float  v8f;

#define Bc 1024
#define Tc 128
#define Hc 768
#define HP 792              // padded LDS row stride (bf16): 12-bank row skew -> conflict-free A fetch
#define NT 6                // N tiles per wave (48 tiles / 8 waves)

__device__ __forceinline__ v8f vzero8() {
    v8f z;
#pragma unroll
    for (int i = 0; i < 8; ++i) z[i] = 0.0f;
    return z;
}

// WGP-scope prefetch: default (unprinted) scope field == 0 == SCOPE_WGP,
// i.e. "pull into all cache levels on miss; stop on WGP hit".
__device__ __forceinline__ void prefetch_wgp(const void* p) {
    asm volatile("global_prefetch_b8 %0, off" :: "v"(p));
}

// A operand: 16x32 bf16 tile from LDS (ISA layout):
//   lane l: m = l&15; chunk0 at k + 8*(l>>4) -> VGPR0-3, chunk1 at +16 -> VGPR4-7
__device__ __forceinline__ v16bf load_a_lds(const __bf16* ldsbase, int m, int hi, int kb) {
    const __bf16* p = ldsbase + m * HP + kb + hi * 8;
    v8bf lo = *(const v8bf*)(p);
    v8bf hh = *(const v8bf*)(p + 16);
    v16bf r;
#pragma unroll
    for (int i = 0; i < 8; ++i) { r[i] = lo[i]; r[i + 8] = hh[i]; }
    return r;
}

// B operand: 32x16 bf16 tile; column n == contiguous bf16 weight row n:
//   lane l: n = l&15; chunk0 at k + 8*(l>>4) -> VGPR0-3, chunk1 at +16 -> VGPR4-7
__device__ __forceinline__ v16bf load_b_glb(const __bf16* rowptr, int hi, int kb) {
    const __bf16* p = rowptr + kb + hi * 8;
    v8bf lo = *(const v8bf*)(p);
    v8bf hh = *(const v8bf*)(p + 16);
    v16bf r;
#pragma unroll
    for (int i = 0; i < 8; ++i) { r[i] = lo[i]; r[i + 8] = hh[i]; }
    return r;
}

__device__ __forceinline__ v8f wmma_bf16(v16bf a, v16bf b, v8f c) {
    return __builtin_amdgcn_wmma_f32_16x16x32_bf16(
        false, a, false, b, (short)0, c, false, false);
}

// Full [16 x Hc] x [Hc x 16*NT] GEMM tile-set.  Register footprint kept
// spill-free (the manual double-buffer variant spilled); L2 latency is hidden
// by WGP-scope prefetch 2 k-blocks ahead + the compiler's 3-deep load pipe.
__device__ __forceinline__ void gemm_tiles(const __bf16* aLds,
                                           const __bf16* const* rowp,
                                           int nl, int hi, v8f* acc) {
    for (int kb = 0; kb < Hc; kb += 32) {
        v16bf a = load_a_lds(aLds, nl, hi, kb);
#pragma unroll
        for (int j = 0; j < NT; ++j) {
            prefetch_wgp((const void*)(rowp[j] + kb + 64 + hi * 16));
            v16bf b = load_b_glb(rowp[j], hi, kb);
            acc[j] = wmma_bf16(a, b, acc[j]);
        }
    }
}

// ---------------------------------------------------------------------------
// One-time weight conversion fp32 -> bf16 into workspace:
//   [0)        ga_w_bf16    768*768 bf16
//   [1179648)  gate_wh_bf16 768*768 bf16  (cols 0..767 of gate_w)
//   [2359296)  gate_wx_f32  768 f32       (col 768 of gate_w)
// ---------------------------------------------------------------------------
__global__ void GVGRU_convert(const float* __restrict__ ga_w,
                              const float* __restrict__ gate_w,
                              __bf16* __restrict__ ga_bf,
                              __bf16* __restrict__ gw_bf,
                              float* __restrict__ wx) {
    int i = blockIdx.x * blockDim.x + threadIdx.x;
    if (i < Hc * Hc) {
        ga_bf[i] = (__bf16)ga_w[i];
        int r = i / Hc;
        int c = i - r * Hc;
        gw_bf[i] = (__bf16)gate_w[r * (Hc + 1) + c];
        if (i < Hc) wx[i] = gate_w[i * (Hc + 1) + Hc];
    }
}

// ---------------------------------------------------------------------------
// Main persistent recurrence kernel: grid = 64 blocks (16 batch rows each),
// block = 256 threads = 8 waves; wave w owns N-tiles [w*6, w*6+6).
// ---------------------------------------------------------------------------
__global__ __launch_bounds__(256, 1)
void GVGRU_kernel(const float* __restrict__ x,
                  const float* __restrict__ gate_b,
                  const float* __restrict__ ga_b,
                  const float* __restrict__ out_w,
                  const float* __restrict__ out_b,
                  const __bf16* __restrict__ ga_bf,
                  const __bf16* __restrict__ gw_bf,
                  const float* __restrict__ wx,
                  float* __restrict__ out) {
    __shared__ __bf16 sh_h[16 * HP];    // hidden state (A operand, GEMM1)
    __shared__ __bf16 sh_gh[16 * HP];   // g * h       (A operand, GEMM2)
    __shared__ float  sh_xt[16];        // x[:, T-1-t] for this batch tile

    const int tid   = threadIdx.x;
    const int lane  = tid & 31;
    const int wv    = tid >> 5;
    const int nl    = lane & 15;
    const int hi    = lane >> 4;
    const int bbase = blockIdx.x * 16;

    for (int i = tid; i < 16 * HP; i += 256) sh_h[i] = (__bf16)0.0f;

    const __bf16* garow[NT];
    const __bf16* gwrow[NT];
    float biasg[NT], biash[NT], wxv[NT];
#pragma unroll
    for (int j = 0; j < NT; ++j) {
        int n = (wv * NT + j) * 16 + nl;
        garow[j] = ga_bf + n * Hc;
        gwrow[j] = gw_bf + n * Hc;
        biasg[j] = ga_b[n];
        biash[j] = gate_b[n];
        wxv[j]   = wx[n];
    }
    __syncthreads();

    for (int t = 0; t < Tc; ++t) {
        if (tid < 16) sh_xt[tid] = x[(bbase + tid) * Tc + (Tc - 1 - t)];

        // ---- GEMM1: g_pre = h @ ga_w^T ----
        v8f acc[NT];
#pragma unroll
        for (int j = 0; j < NT; ++j) acc[j] = vzero8();
        gemm_tiles(sh_h, garow, nl, hi, acc);

        // ---- g = sigmoid(.+bias); stage g*h (bf16); keep g in registers ----
        v8f g[NT];
#pragma unroll
        for (int j = 0; j < NT; ++j) {
            int n = (wv * NT + j) * 16 + nl;
#pragma unroll
            for (int v = 0; v < 8; ++v) {
                int m = v + 8 * hi;
                float gg = 1.0f / (1.0f + __expf(-(acc[j][v] + biasg[j])));
                g[j][v] = gg;
                float hv = (float)sh_h[m * HP + n];
                sh_gh[m * HP + n] = (__bf16)(gg * hv);
            }
        }
        __syncthreads();

        float xtv[8];
#pragma unroll
        for (int v = 0; v < 8; ++v) xtv[v] = sh_xt[v + 8 * hi];

        // ---- GEMM2: cand_pre = (g*h) @ gate_w[:, :H]^T ----
        v8f acc2[NT];
#pragma unroll
        for (int j = 0; j < NT; ++j) acc2[j] = vzero8();
        gemm_tiles(sh_gh, gwrow, nl, hi, acc2);

        // ---- relu candidate, convex blend, write h back ----
#pragma unroll
        for (int j = 0; j < NT; ++j) {
            int n = (wv * NT + j) * 16 + nl;
#pragma unroll
            for (int v = 0; v < 8; ++v) {
                int m = v + 8 * hi;
                float pre  = acc2[j][v] + biash[j] + xtv[v] * wxv[j];
                float hcnd = fmaxf(pre, 0.0f);
                float gg   = g[j][v];
                float hold = (float)sh_h[m * HP + n];
                sh_h[m * HP + n] = (__bf16)(gg * hcnd + (1.0f - gg) * hold);
            }
        }
        __syncthreads();
    }

    // ---- output projection: out[b] = h[b,:] . out_w + out_b ----
#pragma unroll
    for (int mi = 0; mi < 2; ++mi) {
        int m = wv + 8 * mi;
        float s = 0.0f;
        for (int n = lane; n < Hc; n += 32)
            s += (float)sh_h[m * HP + n] * out_w[n];
#pragma unroll
        for (int off = 16; off > 0; off >>= 1)
            s += __shfl_down(s, off, 32);
        if (lane == 0) out[bbase + m] = s + out_b[0];
    }
}

// ---------------------------------------------------------------------------
extern "C" void kernel_launch(void* const* d_in, const int* in_sizes, int n_in,
                              void* d_out, int out_size, void* d_ws, size_t ws_size,
                              hipStream_t stream) {
    const float* x      = (const float*)d_in[0];   // [B, T]
    const float* gate_w = (const float*)d_in[1];   // [H, H+1]
    const float* gate_b = (const float*)d_in[2];   // [H]
    const float* ga_w   = (const float*)d_in[3];   // [H, H]
    const float* ga_b   = (const float*)d_in[4];   // [H]
    const float* out_w  = (const float*)d_in[5];   // [1, H]
    const float* out_b  = (const float*)d_in[6];   // [1]
    float* out = (float*)d_out;                    // [B, 1]

    char* ws = (char*)d_ws;
    __bf16* ga_bf = (__bf16*)(ws);
    __bf16* gw_bf = (__bf16*)(ws + (size_t)Hc * Hc * sizeof(__bf16));
    float*  wxcol = (float*) (ws + (size_t)2 * Hc * Hc * sizeof(__bf16));

    {
        int n = Hc * Hc;
        GVGRU_convert<<<(n + 255) / 256, 256, 0, stream>>>(ga_w, gate_w, ga_bf, gw_bf, wxcol);
    }

    GVGRU_kernel<<<Bc / 16, 256, 0, stream>>>(x, gate_b, ga_b, out_w, out_b,
                                              ga_bf, gw_bf, wxcol, out);
}
